// CrossLinear_14628658610264
// MI455X (gfx1250) — compile-verified, matrix operand order checked by reading, not run on hardware
//
#include <hip/hip_runtime.h>
#include <hip/hip_bf16.h>

#define B_ 8
#define T_ 2048
#define C_ 1024
#define SROW 2052   // padded LDS row stride (floats): 4-bank skew per row

typedef __attribute__((ext_vector_type(16))) __bf16 bf16x16;
typedef __attribute__((ext_vector_type(8)))  float  f32x8;

union FragB16 {
    bf16x16 v;
    unsigned short s[16];
    uint4 q[2];
};

__device__ __forceinline__ unsigned short f2bf(float f) {
    unsigned int u = __float_as_uint(f);
    if ((u & 0x7f800000u) == 0x7f800000u) return (unsigned short)(u >> 16); // inf/nan
    u += 0x7fffu + ((u >> 16) & 1u);  // round to nearest even
    return (unsigned short)(u >> 16);
}

__device__ __forceinline__ f32x8 wmma_bf16(bf16x16 a, bf16x16 b, f32x8 c) {
    return __builtin_amdgcn_wmma_f32_16x16x32_bf16(
        /*neg_a=*/false, a, /*neg_b=*/false, b,
        /*c_mod=*/(short)0, c, /*reuse_a=*/false, /*reuse_b=*/false);
}

// ---------------- Kernel 1a: x (f32) -> xbf (bf16) ----------------
__global__ void cvt_x_kernel(const float* __restrict__ x,
                             unsigned short* __restrict__ xbf, int n) {
    int i = blockIdx.x * blockDim.x + threadIdx.x;
    if (i < n) xbf[i] = f2bf(x[i]);
}

// ---------------- Kernel 1b: W (f32, [c,d]) -> WT (bf16, [d,c]) ----------------
__global__ void cvt_wt_kernel(const float* __restrict__ W,
                              unsigned short* __restrict__ WT) {
    int i = blockIdx.x * blockDim.x + threadIdx.x;   // over C*C
    int c = i / C_;
    int d = i % C_;
    WT[(size_t)d * C_ + c] = f2bf(W[i]);
}

// ---------------- Kernel 2: xxT[b,d,s] = (x @ W)^T, bf16 ----------------
// One wave per 32(d) x 32(s) macro-tile (2x2 WMMA tiles); K loop over c.
// 8 b128 loads feed 4 WMMAs per k-step (each fragment reused twice).
__global__ __launch_bounds__(256)
void proj_gemm_kernel(const unsigned short* __restrict__ xbf,
                      const unsigned short* __restrict__ WT,
                      unsigned short* __restrict__ xxT) {
    const int lane = threadIdx.x & 31;
    const int wave = threadIdx.x >> 5;
    const int mn = lane & 15;
    const int hi = lane >> 4;

    const int ST = T_ / 32;   // 64 s macro-tiles
    const int DT = C_ / 32;   // 32 d macro-tiles
    int tile = blockIdx.x * 8 + wave;
    int b  = tile / (DT * ST);
    int r0 = tile % (DT * ST);
    int d0 = (r0 / ST) * 32;
    int s0 = (r0 % ST) * 32;

    // A = WT (M=d, K=c): lane row d, contiguous K chunks
    const unsigned short* ap0 = WT + (size_t)(d0 + mn) * C_ + hi * 8;
    const unsigned short* ap1 = ap0 + (size_t)16 * C_;
    // B = x^T (K=c, N=s): lane col s -> row of xbf, contiguous in c
    const unsigned short* bp0 = xbf + ((size_t)b * T_ + s0 + mn) * C_ + hi * 8;
    const unsigned short* bp1 = bp0 + (size_t)16 * C_;

    f32x8 acc00 = {0.f,0.f,0.f,0.f,0.f,0.f,0.f,0.f};
    f32x8 acc01 = acc00, acc10 = acc00, acc11 = acc00;

    for (int k = 0; k < C_; k += 32) {
        FragB16 a0, a1, b0, b1;
        a0.q[0] = *(const uint4*)(ap0 + k);
        a0.q[1] = *(const uint4*)(ap0 + k + 16);
        a1.q[0] = *(const uint4*)(ap1 + k);
        a1.q[1] = *(const uint4*)(ap1 + k + 16);
        b0.q[0] = *(const uint4*)(bp0 + k);
        b0.q[1] = *(const uint4*)(bp0 + k + 16);
        b1.q[0] = *(const uint4*)(bp1 + k);
        b1.q[1] = *(const uint4*)(bp1 + k + 16);
        acc00 = wmma_bf16(a0.v, b0.v, acc00);
        acc01 = wmma_bf16(a0.v, b1.v, acc01);
        acc10 = wmma_bf16(a1.v, b0.v, acc10);
        acc11 = wmma_bf16(a1.v, b1.v, acc11);
    }

    // D(m=d,n=s): VGPR r -> row (hi*8 + r), lane -> col mn
    unsigned short* o = xxT + ((size_t)b * C_ + d0 + hi * 8) * T_ + s0 + mn;
#pragma unroll
    for (int r = 0; r < 8; ++r) {
        o[(size_t)r * T_]            = f2bf(acc00[r]);
        o[(size_t)r * T_ + 16]       = f2bf(acc01[r]);
        o[(size_t)(r + 16) * T_]      = f2bf(acc10[r]);
        o[(size_t)(r + 16) * T_ + 16] = f2bf(acc11[r]);
    }
}

// ---------------- Kernel 3: fused scores + mask + softmax + attn@xxT ----------------
// Workgroup = 8 waves, owns (b, 16 rows of t). LDS: S[16][SROW] f32.
__global__ __launch_bounds__(256)
void attn_fused_kernel(const unsigned short* __restrict__ xbf,
                       const unsigned char* __restrict__ mask,
                       const unsigned short* __restrict__ xxT,
                       float* __restrict__ out) {
    extern __shared__ float S[];   // 16 * SROW floats = 131,328 bytes

    const int b  = blockIdx.x >> 7;          // T_/16 = 128 row-blocks per batch
    const int t0 = (blockIdx.x & 127) * 16;
    const int lane = threadIdx.x & 31;
    const int wave = threadIdx.x >> 5;
    const int mn = lane & 15;
    const int hi = lane >> 4;

    // ---- Phase 1: S = Q K^T, masked, fp32 into LDS ----
    // wave w covers s in [w*256, w*256+256): 4 groups of 4 s-tiles,
    // A fragment loaded once per k-step and reused across 4 B tiles.
    const unsigned short* qp = xbf + ((size_t)b * T_ + t0 + mn) * C_ + hi * 8;
    for (int sg = 0; sg < 4; ++sg) {
        int s0 = wave * 256 + sg * 64;
        const unsigned short* kp0 =
            xbf + ((size_t)b * T_ + s0 + mn) * C_ + hi * 8;
        f32x8 acc[4];
#pragma unroll
        for (int j = 0; j < 4; ++j)
            acc[j] = (f32x8){0.f,0.f,0.f,0.f,0.f,0.f,0.f,0.f};

        for (int k = 0; k < C_; k += 32) {
            FragB16 a;
            a.q[0] = *(const uint4*)(qp + k);
            a.q[1] = *(const uint4*)(qp + k + 16);
#pragma unroll
            for (int j = 0; j < 4; ++j) {
                const unsigned short* kp = kp0 + (size_t)(j * 16) * C_;
                FragB16 bm;
                bm.q[0] = *(const uint4*)(kp + k);
                bm.q[1] = *(const uint4*)(kp + k + 16);
                acc[j] = wmma_bf16(a.v, bm.v, acc[j]);
            }
        }

#pragma unroll
        for (int j = 0; j < 4; ++j) {
            int s0j = s0 + j * 16;
            const unsigned char* mp =
                mask + ((size_t)b * T_ + t0 + hi * 8) * T_ + s0j + mn;
            float* sp = S + (size_t)(hi * 8) * SROW + s0j + mn;
#pragma unroll
            for (int r = 0; r < 8; ++r) {
                float v = mp[(size_t)r * T_] ? -__builtin_inff() : acc[j][r];
                sp[(size_t)r * SROW] = v;
            }
        }
    }
    __syncthreads();

    // ---- Phase 2: row softmax in LDS (2 rows per wave) ----
#pragma unroll
    for (int rr = 0; rr < 2; ++rr) {
        float* row = S + (size_t)(wave * 2 + rr) * SROW;
        float mx = -__builtin_inff();
        for (int c = lane; c < T_; c += 32) mx = fmaxf(mx, row[c]);
#pragma unroll
        for (int off = 16; off; off >>= 1) mx = fmaxf(mx, __shfl_xor(mx, off, 32));
        float sum = 0.f;
        for (int c = lane; c < T_; c += 32) {
            float e = __expf(row[c] - mx);
            row[c] = e;
            sum += e;
        }
#pragma unroll
        for (int off = 16; off; off >>= 1) sum += __shfl_xor(sum, off, 32);
        float inv = 1.0f / sum;
        for (int c = lane; c < T_; c += 32) row[c] *= inv;
    }
    __syncthreads();

    // ---- Phase 3: out[16, d0:d0+128] = attn @ xxT (K = s) ----
    // A fragment (from LDS, cvt to bf16) reused across 8 B tiles.
    const int d0 = wave * 128;
    f32x8 acc[8];
#pragma unroll
    for (int i = 0; i < 8; ++i)
        acc[i] = (f32x8){0.f,0.f,0.f,0.f,0.f,0.f,0.f,0.f};

    const float* ap0 = S + (size_t)mn * SROW + hi * 8;  // A row = attn row mn
    for (int s0 = 0; s0 < T_; s0 += 32) {
        FragB16 a;
        float4 f0 = *(const float4*)(ap0 + s0);
        float4 f1 = *(const float4*)(ap0 + s0 + 4);
        float4 f2 = *(const float4*)(ap0 + s0 + 16);
        float4 f3 = *(const float4*)(ap0 + s0 + 20);
        a.s[0] = f2bf(f0.x);  a.s[1] = f2bf(f0.y);  a.s[2] = f2bf(f0.z);  a.s[3] = f2bf(f0.w);
        a.s[4] = f2bf(f1.x);  a.s[5] = f2bf(f1.y);  a.s[6] = f2bf(f1.z);  a.s[7] = f2bf(f1.w);
        a.s[8] = f2bf(f2.x);  a.s[9] = f2bf(f2.y);  a.s[10] = f2bf(f2.z); a.s[11] = f2bf(f2.w);
        a.s[12] = f2bf(f3.x); a.s[13] = f2bf(f3.y); a.s[14] = f2bf(f3.z); a.s[15] = f2bf(f3.w);
#pragma unroll
        for (int nt = 0; nt < 8; ++nt) {
            const unsigned short* bp =
                xxT + ((size_t)b * C_ + d0 + nt * 16 + mn) * T_ + s0 + hi * 8;
            FragB16 bm;
            bm.q[0] = *(const uint4*)(bp);
            bm.q[1] = *(const uint4*)(bp + 16);
            acc[nt] = wmma_bf16(a.v, bm.v, acc[nt]);
        }
    }

    float* op = out + ((size_t)b * T_ + t0 + hi * 8) * C_ + d0 + mn;
#pragma unroll
    for (int nt = 0; nt < 8; ++nt)
#pragma unroll
        for (int r = 0; r < 8; ++r)
            op[(size_t)r * C_ + nt * 16] = acc[nt][r];
}

extern "C" void kernel_launch(void* const* d_in, const int* in_sizes, int n_in,
                              void* d_out, int out_size, void* d_ws, size_t ws_size,
                              hipStream_t stream) {
    const float*         x    = (const float*)d_in[0];
    const unsigned char* mask = (const unsigned char*)d_in[1];  // jnp.bool_ -> byte
    const float*         W    = (const float*)d_in[2];
    float*               out  = (float*)d_out;

    // Workspace layout (bf16 halves): xbf | WT | xxT  (~66 MB total)
    unsigned short* xbf = (unsigned short*)d_ws;
    unsigned short* WT  = xbf + (size_t)B_ * T_ * C_;
    unsigned short* xxT = WT  + (size_t)C_ * C_;

    const int nx = B_ * T_ * C_;                 // 16,777,216
    cvt_x_kernel<<<nx / 256, 256, 0, stream>>>(x, xbf, nx);
    cvt_wt_kernel<<<(C_ * C_) / 256, 256, 0, stream>>>(W, WT);

    const int tiles = B_ * (C_ / 32) * (T_ / 32); // 16,384 waves -> 8 waves/block
    proj_gemm_kernel<<<tiles / 8, 256, 0, stream>>>(xbf, WT, xxT);

    const size_t lds_bytes = (size_t)16 * SROW * sizeof(float); // 131,328 B
    attn_fused_kernel<<<B_ * (T_ / 16), 256, lds_bytes, stream>>>(xbf, mask, xxT, out);
}
